// PresGAN_21895743275432
// MI455X (gfx1250) — compile-verified, matrix operand order checked by reading, not run on hardware
//
#include <hip/hip_runtime.h>
#include <hip/hip_bf16.h>
#include <stdint.h>

typedef __attribute__((ext_vector_type(16))) _Float16 v16h;
typedef __attribute__((ext_vector_type(8)))  _Float16 v8h;
typedef __attribute__((ext_vector_type(8)))  float    v8f;

// ---------------- RNG helpers (deterministic, device-side) ----------------
__device__ __forceinline__ unsigned long long sm64(unsigned long long z) {
  z += 0x9E3779B97F4A7C15ULL;
  z = (z ^ (z >> 30)) * 0xBF58476D1CE4E5B9ULL;
  z = (z ^ (z >> 27)) * 0x94D049BB133111EBULL;
  return z ^ (z >> 31);
}
__device__ __forceinline__ float u01(unsigned long long h) {
  return ((float)(unsigned)(h >> 40) + 0.5f) * (1.0f / 16777216.0f);
}

// Assemble a v16h WMMA fragment from two contiguous 8-half (16B) runs.
// Per the CDNA5 16-bit fragment layout, within a 32-wide K chunk a lane needs
// K = {half*8 + 0..7} (elements 0..7) and K = {16 + half*8 + 0..7} (elements 8..15).
__device__ __forceinline__ v16h frag_ld(const _Float16* p_lo, const _Float16* p_hi,
                                        bool valid) {
  const v8h z8 = {};
  v8h lo = valid ? *(const v8h*)p_lo : z8;
  v8h hi = valid ? *(const v8h*)p_hi : z8;
  return __builtin_shufflevector(lo, hi, 0, 1, 2, 3, 4, 5, 6, 7,
                                 8, 9, 10, 11, 12, 13, 14, 15);
}

// ---------------- Layer 1: eps[512,100] x W1[100, 256*16] (K padded to 128) ----
__global__ __launch_bounds__(32) void gemm_l1_kernel(
    const _Float16* __restrict__ eps16,   // [512][128]  (zero-padded K)
    const _Float16* __restrict__ w1t,     // [4096][128] (transposed, padded)
    float* __restrict__ out)              // [512][256][4][4]
{
  const int lane = threadIdx.x & 31;
  const int half = lane >> 4;
  const int l15  = lane & 15;
  const int colt = blockIdx.x;   // 0..255  (4096/16)
  const int rowt = blockIdx.y;   // 0..31   (512/16)
  const int ncol = colt * 16 + l15;
  const int nrowA = rowt * 16 + l15;

  const _Float16* arow = eps16 + (long)nrowA * 128;
  const _Float16* brow = w1t + (long)ncol * 128;

  v8f acc = {};
  for (int k0 = 0; k0 < 128; k0 += 32) {
    const int o = k0 + half * 8;
    v16h a = frag_ld(arow + o, arow + o + 16, true);
    v16h b = frag_ld(brow + o, brow + o + 16, true);
    acc = __builtin_amdgcn_wmma_f32_16x16x32_f16(false, a, false, b,
                                                 (short)0, acc, false, false);
  }
  const int co  = ncol >> 4;
  const int pix = ncol & 15;
#pragma unroll
  for (int v = 0; v < 8; ++v) {
    const int nr = rowt * 16 + v + half * 8;
    out[((long)nr * 256 + co) * 16 + pix] = acc[v];
  }
}

// ---------------- ConvTranspose2d(k=4,s=2,p=1) as parity-class implicit GEMM ---
// act is channels-last: [512][Hin][Hin][Cin]; weights prepacked [tap][Cout][Cin].
__global__ __launch_bounds__(32) void convt_wmma_kernel(
    const _Float16* __restrict__ act,
    const _Float16* __restrict__ wA,
    float* __restrict__ out,            // [512][Cout][2*Hin][2*Hin] (NCHW)
    int Cin, int Cout, int Hin)
{
  const int lane = threadIdx.x & 31;
  const int half = lane >> 4;
  const int l15  = lane & 15;
  const int Hout = Hin * 2;
  const int HH   = Hin * Hin;
  const int tpc  = HH >> 4;          // 16-pixel N-tiles per parity class

  int tile = blockIdx.x;             // (n, class, tile-in-class)
  const int n   = tile / (4 * tpc);
  int rem = tile - n * (4 * tpc);
  const int cls = rem / tpc;
  const int twc = rem - cls * tpc;
  const int py = cls >> 1, px = cls & 1;

  const int idx = twc * 16 + l15;    // this lane's pixel within the class
  const int qy = idx / Hin, qx = idx - qy * Hin;
  const int oy = 2 * qy + py, ox = 2 * qx + px;

  const int co_tile = blockIdx.y;
  const int coA = co_tile * 16 + l15;
  const bool coOK = (coA < Cout);

  v8f acc = {};
  for (int t = 0; t < 4; ++t) {
    const int dy = t >> 1, dx = t & 1;
    const int ky = (1 - py) + 2 * dy;
    const int kx = (1 - px) + 2 * dx;
    const int iy = qy + py - dy;
    const int ix = qx + px - dx;
    const bool valid = (iy >= 0) && (iy < Hin) && (ix >= 0) && (ix < Hin);

    const _Float16* wrow = wA + ((long)(ky * 4 + kx) * Cout + (coOK ? coA : 0)) * Cin;
    const _Float16* arow = act + (((long)n * Hin + (valid ? iy : 0)) * Hin
                                  + (valid ? ix : 0)) * Cin;

    for (int ci0 = 0; ci0 < Cin; ci0 += 32) {
      const int o = ci0 + half * 8;
      __builtin_prefetch(arow + o + 32, 0, 1);   // global_prefetch (next K chunk)
      v16h a = frag_ld(wrow + o, wrow + o + 16, coOK);
      v16h b = frag_ld(arow + o, arow + o + 16, valid);
      acc = __builtin_amdgcn_wmma_f32_16x16x32_f16(false, a, false, b,
                                                   (short)0, acc, false, false);
    }
  }
#pragma unroll
  for (int v = 0; v < 8; ++v) {
    const int co = co_tile * 16 + v + half * 8;
    if (co < Cout)
      out[(((long)n * Cout + co) * Hout + oy) * Hout + ox] = acc[v];
  }
}

// ---------------- BatchNorm (train mode, biased var) ---------------------------
__global__ void bn_reduce_kernel(const float* __restrict__ x,
                                 float* __restrict__ stats, int C, int HW)
{
  __shared__ float ss[256], ss2[256];
  const int c = blockIdx.x, tid = threadIdx.x;
  const int total = 512 * HW;
  float s = 0.f, s2 = 0.f;
  for (int i = tid; i < total; i += 256) {
    const int nn = i / HW, p = i - nn * HW;
    const float v = x[((long)nn * C + c) * HW + p];
    s += v; s2 += v * v;
  }
  ss[tid] = s; ss2[tid] = s2; __syncthreads();
  for (int o = 128; o > 0; o >>= 1) {
    if (tid < o) { ss[tid] += ss[tid + o]; ss2[tid] += ss2[tid + o]; }
    __syncthreads();
  }
  if (tid == 0) {
    const float inv = 1.0f / (float)total;
    const float m = ss[0] * inv;
    const float var = ss2[0] * inv - m * m;
    stats[c] = m;
    stats[C + c] = rsqrtf(var + 1e-5f);
  }
}

// Normalize + ReLU + f16, writing CHANNELS-LAST [n][p][c] for the next ConvT.
__global__ void bn_apply_relu_f16_kernel(const float* __restrict__ x,
                                         const float* __restrict__ g,
                                         const float* __restrict__ b,
                                         const float* __restrict__ stats,
                                         _Float16* __restrict__ out,
                                         int C, int HW, long total)
{
  const long i = (long)blockIdx.x * blockDim.x + threadIdx.x;
  if (i >= total) return;
  const long chw = (long)C * HW;
  const int n = (int)(i / chw);
  const long r = i - (long)n * chw;
  const int c = (int)(r / HW);
  const int p = (int)(r - (long)c * HW);
  float v = (x[i] - stats[c]) * stats[C + c] * g[c] + b[c];
  v = v > 0.f ? v : 0.f;
  out[((long)n * HW + p) * C + c] = (_Float16)v;
}

// ---------------- tanh + residual log-likelihood reduction ---------------------
__global__ void tanh_resid_kernel(const float* __restrict__ pre,
                                  const float* __restrict__ x_tilde,
                                  const float* __restrict__ sigma,
                                  float* __restrict__ lp_x)
{
  __shared__ float ss[256];
  const int n = blockIdx.x, tid = threadIdx.x;
  float acc = 0.f;
  for (int i = tid; i < 3072; i += 256) {
    const int p = i & 1023;                        // sigma is (1,1,32,32)
    const float gv = tanhf(pre[(long)n * 3072 + i]);
    const float r = x_tilde[(long)n * 3072 + i] - gv;
    const float sg = sigma[p];
    acc += r * r / (sg * sg);
  }
  ss[tid] = acc; __syncthreads();
  for (int o = 128; o > 0; o >>= 1) {
    if (tid < o) ss[tid] += ss[tid + o];
    __syncthreads();
  }
  if (tid == 0) lp_x[n] = ss[0];
}

// ---------------- HMC kernels --------------------------------------------------
__global__ void hmc_propose_kernel(const float* __restrict__ eps_cur,
                                   float* __restrict__ eps_prop,
                                   float* __restrict__ p0buf,
                                   float* __restrict__ pfbuf,
                                   const float* __restrict__ step_p,
                                   unsigned long long seed)
{
  const int i = blockIdx.x * blockDim.x + threadIdx.x;
  if (i >= 51200) return;
  const float step = *step_p;
  const float uu1 = u01(sm64(seed ^ (unsigned long long)(2 * i + 1)));
  const float uu2 = u01(sm64(seed ^ (unsigned long long)(2 * i + 2)));
  const float p0 = sqrtf(-2.0f * logf(uu1)) * cosf(6.2831853f * uu2);
  float e = eps_cur[i];
  float p = p0 - step * e * 0.5f;          // grad_U = eps (analytic)
  for (int j = 0; j < 5; ++j) {            // L = 5 leapfrog steps
    e += step * p;
    if (j < 4) p -= step * e;
  }
  p = -(p - step * e * 0.5f);
  eps_prop[i] = e; p0buf[i] = p0; pfbuf[i] = p;
}

__global__ void hmc_reduce_kernel(const float* __restrict__ eps_cur,
                                  const float* __restrict__ eps_prop,
                                  const float* __restrict__ p0buf,
                                  const float* __restrict__ pfbuf,
                                  float* __restrict__ lpec, float* __restrict__ lpep,
                                  float* __restrict__ kc,  float* __restrict__ kp)
{
  __shared__ float s0[128], s1[128], s2[128], s3[128];
  const int n = blockIdx.x, tid = threadIdx.x;
  float a = 0.f, b = 0.f, c = 0.f, d = 0.f;
  for (int j = tid; j < 100; j += 128) {
    const int i = n * 100 + j;
    float x;
    x = eps_cur[i];  a += x * x;
    x = eps_prop[i]; b += x * x;
    x = p0buf[i];    c += x * x;
    x = pfbuf[i];    d += x * x;
  }
  s0[tid] = a; s1[tid] = b; s2[tid] = c; s3[tid] = d; __syncthreads();
  for (int o = 64; o > 0; o >>= 1) {
    if (tid < o) { s0[tid] += s0[tid+o]; s1[tid] += s1[tid+o];
                   s2[tid] += s2[tid+o]; s3[tid] += s3[tid+o]; }
    __syncthreads();
  }
  if (tid == 0) { lpec[n] = s0[0]; lpep[n] = s1[0]; kc[n] = s2[0]; kp[n] = s3[0]; }
}

__global__ void hmc_accept_kernel(const float* __restrict__ lpec,
                                  const float* __restrict__ lpep,
                                  const float* __restrict__ lpxc,
                                  const float* __restrict__ lpxp,
                                  const float* __restrict__ kc,
                                  const float* __restrict__ kp,
                                  float* __restrict__ amask,
                                  float* __restrict__ acc_hist,
                                  float* __restrict__ step_p,
                                  const int* __restrict__ flag_adapt,
                                  int stepidx, int burn_in, unsigned long long seed)
{
  __shared__ float ss[512];
  const int n = threadIdx.x;
  const float cu = 0.5f * (lpec[n] + lpxc[n]);
  const float pu = 0.5f * (lpep[n] + lpxp[n]);
  const float ck = 0.5f * kc[n];
  const float pk = 0.5f * kp[n];
  const float u = u01(sm64(seed ^ (unsigned long long)(n * 1337 + 7)));
  const float a = (u < expf(cu - pu + ck - pk)) ? 1.f : 0.f;
  amask[n] = a;
  acc_hist[stepidx * 512 + n] = a;
  ss[n] = a; __syncthreads();
  for (int o = 256; o > 0; o >>= 1) {
    if (n < o) ss[n] += ss[n + o];
    __syncthreads();
  }
  if (n == 0) {
    const float mean = ss[0] * (1.0f / 512.0f);
    float st = *step_p;
    if (stepidx < burn_in && *flag_adapt == 1)
      st = st + 0.02f * (mean - 0.67f) * st;
    *step_p = st;
  }
}

__global__ void hmc_update_kernel(float* __restrict__ eps_cur,
                                  const float* __restrict__ eps_prop,
                                  const float* __restrict__ amask,
                                  float* __restrict__ out_samples,
                                  int stepidx, int burn_in)
{
  const int i = blockIdx.x * blockDim.x + threadIdx.x;
  if (i >= 51200) return;
  const int n = i / 100;
  const float e = (amask[n] != 0.f) ? eps_prop[i] : eps_cur[i];
  eps_cur[i] = e;
  if (stepidx >= burn_in)
    out_samples[(long)(stepidx - burn_in) * 51200 + i] = e;
}

__global__ void finalize_kernel(const float* __restrict__ acc_hist,
                                const float* __restrict__ step_p,
                                float* __restrict__ out)
{
  const int n = blockIdx.x * blockDim.x + threadIdx.x;
  if (n < 512) {
    float s = 0.f;
    for (int i = 0; i < 4; ++i) s += acc_hist[i * 512 + n];
    out[102400 + n] = s * 0.25f;
  }
  if (n == 0) out[102912] = *step_p;
}

// ---------------- prep kernels -------------------------------------------------
__global__ void init_step_kernel(float* step_p) {
  if (threadIdx.x == 0 && blockIdx.x == 0) *step_p = 0.03f;
}
__global__ void copy_f32_kernel(const float* __restrict__ src,
                                float* __restrict__ dst, long n) {
  const long i = (long)blockIdx.x * blockDim.x + threadIdx.x;
  if (i < n) dst[i] = src[i];
}
// W1 (100,256,4,4) -> f16 transposed [col=co*16+pix][K=128], zero-padded K.
__global__ void w1t_prep_kernel(const float* __restrict__ w1,
                                _Float16* __restrict__ dst) {
  const long i = (long)blockIdx.x * blockDim.x + threadIdx.x;
  if (i >= 4096L * 128) return;
  const int col = (int)(i >> 7);
  const int k = (int)(i & 127);
  dst[i] = (k < 100) ? (_Float16)w1[(long)k * 4096 + col] : (_Float16)0.0f;
}
// Conv weights (Cin,Cout,4,4) f32 -> f16 prepacked [tap][Cout][Cin].
__global__ void wconv_prep_kernel(const float* __restrict__ w,
                                  _Float16* __restrict__ dst,
                                  int Cin, int Cout) {
  const long total = (long)Cin * Cout * 16;
  const long i = (long)blockIdx.x * blockDim.x + threadIdx.x;
  if (i >= total) return;
  const long coci = (long)Cout * Cin;
  const int t = (int)(i / coci);
  const long r = i - (long)t * coci;
  const int co = (int)(r / Cin);
  const int ci = (int)(r - (long)co * Cin);
  dst[i] = (_Float16)w[((long)ci * Cout + co) * 16 + t];
}
__global__ void eps16_prep_kernel(const float* __restrict__ eps,
                                  _Float16* __restrict__ dst) {
  const int i = blockIdx.x * blockDim.x + threadIdx.x;
  if (i >= 512 * 128) return;
  const int n = i >> 7, j = i & 127;
  dst[i] = (j < 100) ? (_Float16)eps[n * 100 + j] : (_Float16)0.0f;
}

// ---------------- host orchestration ------------------------------------------
extern "C" void kernel_launch(void* const* d_in, const int* in_sizes, int n_in,
                              void* d_out, int out_size, void* d_ws, size_t ws_size,
                              hipStream_t stream) {
  (void)in_sizes; (void)n_in; (void)out_size; (void)ws_size;
  const float* x_tilde = (const float*)d_in[0];
  const float* eps_init = (const float*)d_in[1];
  const float* sigma = (const float*)d_in[2];
  const float* w1 = (const float*)d_in[3];
  const float* g1 = (const float*)d_in[4];
  const float* b1 = (const float*)d_in[5];
  const float* w2 = (const float*)d_in[6];
  const float* g2 = (const float*)d_in[7];
  const float* b2 = (const float*)d_in[8];
  const float* w3 = (const float*)d_in[9];
  const float* g3 = (const float*)d_in[10];
  const float* b3 = (const float*)d_in[11];
  const float* w4 = (const float*)d_in[12];
  const int* flag_adapt = (const int*)d_in[16];
  float* out = (float*)d_out;

  char* ws = (char*)d_ws;
  size_t cur = 0;
  auto alloc = [&](size_t bytes) -> void* {
    cur = (cur + 255) & ~(size_t)255;
    void* p = ws + cur;
    cur += bytes;
    return p;
  };

  float* eps_cur  = (float*)alloc(51200 * 4);
  float* eps_prop = (float*)alloc(51200 * 4);
  float* p0buf    = (float*)alloc(51200 * 4);
  float* pfbuf    = (float*)alloc(51200 * 4);
  _Float16* eps16 = (_Float16*)alloc(512 * 128 * 2);
  _Float16* w1t   = (_Float16*)alloc(4096L * 128 * 2);
  _Float16* w2p   = (_Float16*)alloc(16L * 128 * 256 * 2);
  _Float16* w3p   = (_Float16*)alloc(16L * 64 * 128 * 2);
  _Float16* w4p   = (_Float16*)alloc(16L * 3 * 64 * 2);
  float* pre32    = (float*)alloc(512L * 64 * 256 * 4);     // max pre-BN tensor
  _Float16* act16A = (_Float16*)alloc(512L * 64 * 256 * 2); // l1/l3 activations (CL)
  _Float16* act16B = (_Float16*)alloc(512L * 128 * 64 * 2); // l2 activations (CL)
  float* stats = (float*)alloc(2 * 256 * 4);
  float* lpec = (float*)alloc(512 * 4);
  float* lpep = (float*)alloc(512 * 4);
  float* kc   = (float*)alloc(512 * 4);
  float* kp   = (float*)alloc(512 * 4);
  float* lpxc = (float*)alloc(512 * 4);
  float* lpxp = (float*)alloc(512 * 4);
  float* amask = (float*)alloc(512 * 4);
  float* acc_hist = (float*)alloc(4 * 512 * 4);
  float* step_p = (float*)alloc(256);

  init_step_kernel<<<1, 32, 0, stream>>>(step_p);
  copy_f32_kernel<<<200, 256, 0, stream>>>(eps_init, eps_cur, 51200);
  w1t_prep_kernel<<<2048, 256, 0, stream>>>(w1, w1t);
  wconv_prep_kernel<<<2048, 256, 0, stream>>>(w2, w2p, 256, 128);
  wconv_prep_kernel<<<512, 256, 0, stream>>>(w3, w3p, 128, 64);
  wconv_prep_kernel<<<12, 256, 0, stream>>>(w4, w4p, 64, 3);

  auto forward = [&](const float* eps_src, float* lp_x) {
    eps16_prep_kernel<<<256, 256, 0, stream>>>(eps_src, eps16);
    // L1: 1x1 -> 4x4 (pure GEMM)
    gemm_l1_kernel<<<dim3(256, 32), 32, 0, stream>>>(eps16, w1t, pre32);
    bn_reduce_kernel<<<256, 256, 0, stream>>>(pre32, stats, 256, 16);
    bn_apply_relu_f16_kernel<<<8192, 256, 0, stream>>>(pre32, g1, b1, stats,
                                                       act16A, 256, 16, 2097152L);
    // L2: 4 -> 8
    convt_wmma_kernel<<<dim3(2048, 8), 32, 0, stream>>>(act16A, w2p, pre32, 256, 128, 4);
    bn_reduce_kernel<<<128, 256, 0, stream>>>(pre32, stats, 128, 64);
    bn_apply_relu_f16_kernel<<<16384, 256, 0, stream>>>(pre32, g2, b2, stats,
                                                        act16B, 128, 64, 4194304L);
    // L3: 8 -> 16
    convt_wmma_kernel<<<dim3(8192, 4), 32, 0, stream>>>(act16B, w3p, pre32, 128, 64, 8);
    bn_reduce_kernel<<<64, 256, 0, stream>>>(pre32, stats, 64, 256);
    bn_apply_relu_f16_kernel<<<32768, 256, 0, stream>>>(pre32, g3, b3, stats,
                                                        act16A, 64, 256, 8388608L);
    // L4: 16 -> 32
    convt_wmma_kernel<<<dim3(32768, 1), 32, 0, stream>>>(act16A, w4p, pre32, 64, 3, 16);
    tanh_resid_kernel<<<512, 256, 0, stream>>>(pre32, x_tilde, sigma, lp_x);
  };

  const int burn_in = 2;
  for (int s = 0; s < 4; ++s) {
    hmc_propose_kernel<<<200, 256, 0, stream>>>(eps_cur, eps_prop, p0buf, pfbuf,
                                                step_p,
                                                0xC0FFEE0000ULL + (unsigned long long)s * 0x9E3779B9ULL);
    hmc_reduce_kernel<<<512, 128, 0, stream>>>(eps_cur, eps_prop, p0buf, pfbuf,
                                               lpec, lpep, kc, kp);
    forward(eps_cur, lpxc);
    forward(eps_prop, lpxp);
    hmc_accept_kernel<<<1, 512, 0, stream>>>(lpec, lpep, lpxc, lpxp, kc, kp,
                                             amask, acc_hist, step_p, flag_adapt,
                                             s, burn_in,
                                             0xBADF00D0000ULL + (unsigned long long)s);
    hmc_update_kernel<<<200, 256, 0, stream>>>(eps_cur, eps_prop, amask, out, s, burn_in);
  }
  finalize_kernel<<<2, 256, 0, stream>>>(acc_hist, step_p, out);
}
